// SalesLSTM_29824252903914
// MI455X (gfx1250) — matrix operand order repaired and checked by reading, not compiled, not run on hardware
//
#include <hip/hip_runtime.h>
#include <hip/hip_bf16.h>
#include <math.h>

// ---- CDNA5 WMMA vector types --------------------------------------------
typedef __attribute__((ext_vector_type(16))) _Float16 v16h;
typedef __attribute__((ext_vector_type(8)))  _Float16 v8h;
typedef __attribute__((ext_vector_type(8)))  float    v8f;

#define Hh   64      // hidden size
#define G4   256     // 4*H gate columns
#define GS   260     // padded LDS gate-row stride (floats): 8*GS % 64 != 0
#define OUTD 7

#define LOG2E  1.44269504088896340736f

// Branch-free activations on raw CDNA5 transcendentals (v_exp_f32 / v_rcp_f32).
__device__ __forceinline__ float fast_sigmoid(float x) {
    float e = __builtin_amdgcn_exp2f(x * -LOG2E);      // 2^(-x*log2e) = e^-x
    return __builtin_amdgcn_rcpf(1.0f + e);
}
__device__ __forceinline__ float fast_tanh(float x) {
    float e = __builtin_amdgcn_exp2f(x * (-2.0f * LOG2E));  // e^-2x
    float r = __builtin_amdgcn_rcpf(1.0f + e);              // sigmoid(2x)
    return __builtin_fmaf(2.0f, r, -1.0f);
}

// Fused 2-layer LSTM: one workgroup = 16 batch rows, 8 wave32, all T steps.
__global__ __launch_bounds__(256, 1)
void lstm2_fused_kernel(const float* __restrict__ x,      // [B,T,1]
                        const float* __restrict__ Wih0,   // [256,1]
                        const float* __restrict__ Whh0,   // [256,64]
                        const float* __restrict__ bih0,   // [256]
                        const float* __restrict__ bhh0,   // [256]
                        const float* __restrict__ Wih1,   // [256,64]
                        const float* __restrict__ Whh1,   // [256,64]
                        const float* __restrict__ bih1,   // [256]
                        const float* __restrict__ bhh1,   // [256]
                        const float* __restrict__ Wfc,    // [7,64]
                        const float* __restrict__ bfc,    // [7]
                        float* __restrict__ out,          // [B,7]
                        int Tdim)
{
    __shared__ __align__(16) _Float16 h1buf[16 * Hh];   // layer-0 h state / A for layer-1 input GEMM
    __shared__ __align__(16) _Float16 h2buf[16 * Hh];   // layer-1 h state
    __shared__ __align__(16) float    Gbuf[16 * GS];    // gate scratch [16][256] padded
    __shared__ float xbuf[16];

    const int tid    = threadIdx.x;
    const int wave   = tid >> 5;
    const int lane   = tid & 31;
    const int laneHi = lane >> 4;     // 0/1 half-wave
    const int laneLo = lane & 15;
    const int b0     = blockIdx.x * 16;

    // ---- pin B-fragments (f32 -> f16) in registers: contiguous k order (k = 16*laneHi + j) ----
    // B[k][n]: lane = (n%16) + 16*(K/16 within 32-K frag), half j -> K = 32*kf + 16*laneHi + j
    v16h Bw0[2][2], Bi1[2][2], Bw1[2][2];
    float wih0v[2], bias0v[2], bias1v[2];
    #pragma unroll
    for (int tile = 0; tile < 2; ++tile) {
        const int n = wave * 32 + tile * 16 + laneLo;   // gate column
        wih0v[tile]  = Wih0[n];
        bias0v[tile] = bih0[n] + bhh0[n];
        bias1v[tile] = bih1[n] + bhh1[n];
        #pragma unroll
        for (int kf = 0; kf < 2; ++kf) {
            const int kb = kf * 32 + laneHi * 16;
            v16h bv0, bv1, bv2;
            #pragma unroll
            for (int j = 0; j < 16; ++j) {
                bv0[j] = (_Float16)Whh0[n * Hh + kb + j];
                bv1[j] = (_Float16)Wih1[n * Hh + kb + j];
                bv2[j] = (_Float16)Whh1[n * Hh + kb + j];
            }
            Bw0[tile][kf] = bv0; Bi1[tile][kf] = bv1; Bw1[tile][kf] = bv2;
        }
    }

    // zero h state; stage x_0
    for (int i = tid; i < 16 * Hh; i += 256) {
        h1buf[i] = (_Float16)0.f;
        h2buf[i] = (_Float16)0.f;
    }
    if (tid < 16) xbuf[tid] = x[(size_t)(b0 + tid) * Tdim];

    // per-thread c-state: 4 (row, col) pairs; row = erow0 + 4k, col = ecol
    float c0s[4] = {0.f, 0.f, 0.f, 0.f};
    float c1s[4] = {0.f, 0.f, 0.f, 0.f};
    const int erow0 = tid >> 6;
    const int ecol  = tid & 63;

    // A-fragment loader from LDS h-state (16x32 A, contiguous k order k = 16*laneHi + j):
    // lane row = laneLo, halves 0..15 <- K block 16*laneHi (contiguous, matches B)
    auto loadA = [&](const _Float16* hb, int kf) -> v16h {
        const _Float16* p = hb + laneLo * Hh + kf * 32 + laneHi * 16;
        v8h lo = *(const v8h*)(p);
        v8h hi = *(const v8h*)(p + 8);
        return __builtin_shufflevector(lo, hi, 0,1,2,3,4,5,6,7,8,9,10,11,12,13,14,15);
    };

    __syncthreads();

    for (int t = 0; t < Tdim; ++t) {
        // ---- layer 0: gates = x_t*Wih0 + b + h1 @ Whh0^T  (WMMA) ----
        {
            v16h a0 = loadA(h1buf, 0);
            v16h a1 = loadA(h1buf, 1);
            #pragma unroll
            for (int tile = 0; tile < 2; ++tile) {
                const int Ncol = wave * 32 + tile * 16 + laneLo;
                v8f acc;
                #pragma unroll
                for (int r = 0; r < 8; ++r)
                    acc[r] = xbuf[r + laneHi * 8] * wih0v[tile] + bias0v[tile];
                acc = __builtin_amdgcn_wmma_f32_16x16x32_f16(false, a0, false, Bw0[tile][0],
                                                             (short)0, acc, false, false);
                acc = __builtin_amdgcn_wmma_f32_16x16x32_f16(false, a1, false, Bw0[tile][1],
                                                             (short)0, acc, false, false);
                #pragma unroll
                for (int r = 0; r < 8; ++r)
                    Gbuf[(r + laneHi * 8) * GS + Ncol] = acc[r];
            }
        }
        __syncthreads();

        // ---- layer 0 elementwise: i,f,g,o -> c,h ----
        #pragma unroll
        for (int k = 0; k < 4; ++k) {
            const int row = erow0 + 4 * k;
            const float gi = Gbuf[row * GS + ecol];
            const float gf = Gbuf[row * GS +  64 + ecol];
            const float gg = Gbuf[row * GS + 128 + ecol];
            const float go = Gbuf[row * GS + 192 + ecol];
            const float i_ = fast_sigmoid(gi);
            const float f_ = fast_sigmoid(gf);
            const float g_ = fast_tanh(gg);
            const float o_ = fast_sigmoid(go);
            const float c  = f_ * c0s[k] + i_ * g_;
            c0s[k] = c;
            h1buf[row * Hh + ecol] = (_Float16)(o_ * fast_tanh(c));
        }
        __syncthreads();

        // ---- layer 1: gates = b + h1_t @ Wih1^T + h2 @ Whh1^T  (WMMA x4) ----
        {
            v16h a10 = loadA(h1buf, 0);
            v16h a11 = loadA(h1buf, 1);
            v16h a20 = loadA(h2buf, 0);
            v16h a21 = loadA(h2buf, 1);
            #pragma unroll
            for (int tile = 0; tile < 2; ++tile) {
                const int Ncol = wave * 32 + tile * 16 + laneLo;
                v8f acc;
                #pragma unroll
                for (int r = 0; r < 8; ++r) acc[r] = bias1v[tile];
                acc = __builtin_amdgcn_wmma_f32_16x16x32_f16(false, a10, false, Bi1[tile][0],
                                                             (short)0, acc, false, false);
                acc = __builtin_amdgcn_wmma_f32_16x16x32_f16(false, a11, false, Bi1[tile][1],
                                                             (short)0, acc, false, false);
                acc = __builtin_amdgcn_wmma_f32_16x16x32_f16(false, a20, false, Bw1[tile][0],
                                                             (short)0, acc, false, false);
                acc = __builtin_amdgcn_wmma_f32_16x16x32_f16(false, a21, false, Bw1[tile][1],
                                                             (short)0, acc, false, false);
                #pragma unroll
                for (int r = 0; r < 8; ++r)
                    Gbuf[(r + laneHi * 8) * GS + Ncol] = acc[r];
            }
        }
        __syncthreads();

        // ---- layer 1 elementwise (+ stage x_{t+1} for next step) ----
        if (tid < 16 && t + 1 < Tdim)
            xbuf[tid] = x[(size_t)(b0 + tid) * Tdim + t + 1];
        #pragma unroll
        for (int k = 0; k < 4; ++k) {
            const int row = erow0 + 4 * k;
            const float gi = Gbuf[row * GS + ecol];
            const float gf = Gbuf[row * GS +  64 + ecol];
            const float gg = Gbuf[row * GS + 128 + ecol];
            const float go = Gbuf[row * GS + 192 + ecol];
            const float i_ = fast_sigmoid(gi);
            const float f_ = fast_sigmoid(gf);
            const float g_ = fast_tanh(gg);
            const float o_ = fast_sigmoid(go);
            const float c  = f_ * c1s[k] + i_ * g_;
            c1s[k] = c;
            h2buf[row * Hh + ecol] = (_Float16)(o_ * fast_tanh(c));
        }
        __syncthreads();
    }

    // ---- final FC: out[b,:] = h2_last @ Wfc^T + bfc ----
    if (tid < 16 * OUTD) {
        const int row = tid / OUTD;
        const int o   = tid % OUTD;
        float acc = bfc[o];
        #pragma unroll
        for (int k = 0; k < Hh; ++k)
            acc += (float)h2buf[row * Hh + k] * Wfc[o * Hh + k];
        out[(size_t)(b0 + row) * OUTD + o] = acc;
    }
}

extern "C" void kernel_launch(void* const* d_in, const int* in_sizes, int n_in,
                              void* d_out, int out_size, void* d_ws, size_t ws_size,
                              hipStream_t stream) {
    const float* x    = (const float*)d_in[0];
    const float* Wih0 = (const float*)d_in[1];
    const float* Whh0 = (const float*)d_in[2];
    const float* bih0 = (const float*)d_in[3];
    const float* bhh0 = (const float*)d_in[4];
    const float* Wih1 = (const float*)d_in[5];
    const float* Whh1 = (const float*)d_in[6];
    const float* bih1 = (const float*)d_in[7];
    const float* bhh1 = (const float*)d_in[8];
    const float* Wfc  = (const float*)d_in[9];
    const float* bfc  = (const float*)d_in[10];
    float* out = (float*)d_out;

    const int Tdim = 512;
    const int Bdim = in_sizes[0] / Tdim;          // x is [B,T,1]
    const int nblocks = Bdim / 16;

    lstm2_fused_kernel<<<nblocks, 256, 0, stream>>>(
        x, Wih0, Whh0, bih0, bhh0, Wih1, Whh1, bih1, bhh1, Wfc, bfc, out, Tdim);
}
